// DeformConv_3891240370677
// MI455X (gfx1250) — matrix operand order repaired and checked by reading, not compiled, not run on hardware
//
#include <hip/hip_runtime.h>

typedef __attribute__((ext_vector_type(16))) __bf16 v16bf;
typedef __attribute__((ext_vector_type(2)))  __bf16 v2bf;
typedef __attribute__((ext_vector_type(8)))  float  v8f;

#define BATCH 8
#define CIN   128
#define COUT  128
#define HH    64
#define WW    64
#define HW    (HH * WW)
#define KK    9
#define BSTR  136   // LDS stride (bf16 elems): 272B = 17*16B
#define OSTR  72    // LDS stride (f32) for offset rows

union Frag { v16bf v; uint4 q[2]; };
union U4   { uint4 v; unsigned a[4]; };
union U4B  { uint4 v; unsigned a[4]; v2bf h[4]; };

__device__ __forceinline__ unsigned short f2bf(float f) {
    union { float f; unsigned u; } cv; cv.f = f;
    unsigned u = cv.u;
    unsigned r = u + 0x7FFFu + ((u >> 16) & 1u);   // RNE
    return (unsigned short)(r >> 16);
}

__device__ __forceinline__ unsigned pack_bf16(float a, float b) {
#if defined(__has_builtin)
#if __has_builtin(__builtin_amdgcn_cvt_pk_bf16_f32)
    {
        auto r = __builtin_amdgcn_cvt_pk_bf16_f32(a, b);
        if constexpr (sizeof(r) == 4) {
            union { decltype(r) v; unsigned u; } cv; cv.v = r; return cv.u;
        } else {
            return (unsigned)f2bf(a) | ((unsigned)f2bf(b) << 16);
        }
    }
#else
    return (unsigned)f2bf(a) | ((unsigned)f2bf(b) << 16);
#endif
#else
    return (unsigned)f2bf(a) | ((unsigned)f2bf(b) << 16);
#endif
}

__device__ __forceinline__ int iclamp(int v, int lo, int hi) {
    return v < lo ? lo : (v > hi ? hi : v);
}

// -------- prep: weights -> bf16 [kk][co][c] --------
__global__ __launch_bounds__(256)
void prep_w(const float* __restrict__ w, const float* __restrict__ woff,
            unsigned short* __restrict__ w_main,
            unsigned short* __restrict__ w_offp) {
    int idx = blockIdx.x * 256 + threadIdx.x;
    if (idx < KK * COUT * CIN) {
        int kk = idx / (COUT * CIN);
        int rem = idx % (COUT * CIN);
        int co = rem / CIN, c = rem % CIN;
        int ky = kk / 3, kx = kk % 3;
        w_main[idx] = f2bf(w[((co * CIN + c) * 3 + ky) * 3 + kx]);
    }
    if (idx < KK * 32 * CIN) {
        int kk = idx / (32 * CIN);
        int rem = idx % (32 * CIN);
        int r = rem / CIN, c = rem % CIN;
        int ky = kk / 3, kx = kk % 3;
        float v = (r < 18) ? woff[((r * CIN + c) * 3 + ky) * 3 + kx] : 0.f;
        w_offp[idx] = f2bf(v);
    }
}

// -------- prep: x f32 NCHW -> bf16 NHWC, LDS-tiled so both sides coalesce ----
__global__ __launch_bounds__(256)
void prep_x(const float* __restrict__ x, unsigned short* __restrict__ xh) {
    __shared__ __align__(16) unsigned short t[64 * BSTR];
    const int y  = blockIdx.x;
    const int b  = blockIdx.y;
    const int p  = threadIdx.x & 63;
    const int cg = threadIdx.x >> 6;
    const float* src = x + ((size_t)(b * CIN + cg * 32) * HH + y) * WW + p;
    #pragma unroll
    for (int i = 0; i < 32; i += 2) {
        float f0 = src[(size_t)i * HW];
        float f1 = src[(size_t)(i + 1) * HW];
        *(unsigned*)&t[p * BSTR + cg * 32 + i] = pack_bf16(f0, f1);
    }
    __syncthreads();
    unsigned short* dst = xh + ((size_t)(b * HH + y) * WW + p) * CIN + cg * 32;
    #pragma unroll
    for (int i = 0; i < 32; i += 8)
        *(uint4*)&dst[i] = *(const uint4*)&t[p * BSTR + cg * 32 + i];
}

// ---------------- staging: NHWC bf16 fast path ----------------
__device__ __forceinline__ void stage_int_h(unsigned short* __restrict__ buf,
                                            const unsigned short* __restrict__ xhb,
                                            int ho, int p, int cg, int ky, int kx) {
    const int y  = ho - 1 + ky;
    const int xc = p - 1 + kx;
    const unsigned msk = ((y >= 0) & (y < HH) & (xc >= 0) & (xc < WW)) ? 0xFFFFFFFFu : 0u;
    const unsigned short* src = xhb
        + ((size_t)(iclamp(y, 0, HH - 1) * WW + iclamp(xc, 0, WW - 1))) * CIN + cg * 32;
    #pragma unroll
    for (int i = 0; i < 32; i += 8) {
        U4 d; d.v = *(const uint4*)&src[i];
        d.a[0] &= msk; d.a[1] &= msk; d.a[2] &= msk; d.a[3] &= msk;
        *(uint4*)&buf[p * BSTR + cg * 32 + i] = d.v;
    }
}

__device__ __forceinline__ void stage_bilin_h(unsigned short* __restrict__ buf,
                                              const unsigned short* __restrict__ xhb,
                                              const float* __restrict__ s_off,
                                              int ho, int p, int cg, int kk,
                                              int ky, int kx) {
    float dy = s_off[(2 * kk + 0) * OSTR + p];
    float dx = s_off[(2 * kk + 1) * OSTR + p];
    float py = (float)(ho - 1 + ky) + dy;
    float px = (float)(p  - 1 + kx) + dx;
    float y0f = floorf(py), x0f = floorf(px);
    float ly = py - y0f, lx = px - x0f;
    int y0 = (int)y0f, x0 = (int)x0f;
    float vy0 = (y0 >= 0 && y0 < HH)         ? 1.f : 0.f;
    float vy1 = (y0 + 1 >= 0 && y0 + 1 < HH) ? 1.f : 0.f;
    float vx0 = (x0 >= 0 && x0 < WW)         ? 1.f : 0.f;
    float vx1 = (x0 + 1 >= 0 && x0 + 1 < WW) ? 1.f : 0.f;
    int yc0 = iclamp(y0, 0, HH - 1), yc1 = iclamp(y0 + 1, 0, HH - 1);
    int xc0 = iclamp(x0, 0, WW - 1), xc1 = iclamp(x0 + 1, 0, WW - 1);
    float W00 = (1.f - ly) * (1.f - lx) * vy0 * vx0;
    float W01 = (1.f - ly) * lx         * vy0 * vx1;
    float W10 = ly         * (1.f - lx) * vy1 * vx0;
    float W11 = ly         * lx         * vy1 * vx1;
    // packed-bf16 tap weights (V_PK_FMA_BF16 path; operand is bf16 anyway)
    const __bf16 b00 = (__bf16)W00, b01 = (__bf16)W01;
    const __bf16 b10 = (__bf16)W10, b11 = (__bf16)W11;
    const v2bf P00 = {b00, b00}, P01 = {b01, b01};
    const v2bf P10 = {b10, b10}, P11 = {b11, b11};
    const unsigned short* a00 = xhb + ((size_t)(yc0 * WW + xc0)) * CIN + cg * 32;
    const unsigned short* a01 = xhb + ((size_t)(yc0 * WW + xc1)) * CIN + cg * 32;
    const unsigned short* a10 = xhb + ((size_t)(yc1 * WW + xc0)) * CIN + cg * 32;
    const unsigned short* a11 = xhb + ((size_t)(yc1 * WW + xc1)) * CIN + cg * 32;
    #pragma unroll
    for (int ch = 0; ch < 32; ch += 8) {
        U4B q00, q01, q10, q11, r;
        q00.v = *(const uint4*)&a00[ch];
        q01.v = *(const uint4*)&a01[ch];
        q10.v = *(const uint4*)&a10[ch];
        q11.v = *(const uint4*)&a11[ch];
        #pragma unroll
        for (int d = 0; d < 4; ++d) {
            v2bf s = q00.h[d] * P00;
            s = q01.h[d] * P01 + s;
            s = q10.h[d] * P10 + s;
            s = q11.h[d] * P11 + s;
            r.h[d] = s;
        }
        *(uint4*)&buf[p * BSTR + cg * 32 + ch] = r.v;
    }
}

// ---------------- staging: NCHW f32 fallback ----------------
__device__ __forceinline__ void stage_int_f(unsigned short* __restrict__ buf,
                                            const float* __restrict__ xb,
                                            int ho, int p, int cg, int ky, int kx) {
    const int y  = ho - 1 + ky;
    const int xc = p - 1 + kx;
    const float okf = ((y >= 0) & (y < HH) & (xc >= 0) & (xc < WW)) ? 1.f : 0.f;
    const float* xrow = xb + iclamp(y, 0, HH - 1) * WW + iclamp(xc, 0, WW - 1)
                           + (size_t)(cg * 32) * HW;
    #pragma unroll
    for (int i = 0; i < 32; i += 2) {
        float f0 = xrow[(size_t)i * HW]       * okf;
        float f1 = xrow[(size_t)(i + 1) * HW] * okf;
        *(unsigned*)&buf[p * BSTR + cg * 32 + i] = pack_bf16(f0, f1);
    }
}

__device__ __forceinline__ void stage_bilin_f(unsigned short* __restrict__ buf,
                                              const float* __restrict__ xb,
                                              const float* __restrict__ s_off,
                                              int ho, int p, int cg, int kk,
                                              int ky, int kx) {
    float dy = s_off[(2 * kk + 0) * OSTR + p];
    float dx = s_off[(2 * kk + 1) * OSTR + p];
    float py = (float)(ho - 1 + ky) + dy;
    float px = (float)(p  - 1 + kx) + dx;
    float y0f = floorf(py), x0f = floorf(px);
    float ly = py - y0f, lx = px - x0f;
    int y0 = (int)y0f, x0 = (int)x0f;
    float vy0 = (y0 >= 0 && y0 < HH)         ? 1.f : 0.f;
    float vy1 = (y0 + 1 >= 0 && y0 + 1 < HH) ? 1.f : 0.f;
    float vx0 = (x0 >= 0 && x0 < WW)         ? 1.f : 0.f;
    float vx1 = (x0 + 1 >= 0 && x0 + 1 < WW) ? 1.f : 0.f;
    int yc0 = iclamp(y0, 0, HH - 1), yc1 = iclamp(y0 + 1, 0, HH - 1);
    int xc0 = iclamp(x0, 0, WW - 1), xc1 = iclamp(x0 + 1, 0, WW - 1);
    float w00 = (1.f - ly) * (1.f - lx) * vy0 * vx0;
    float w01 = (1.f - ly) * lx         * vy0 * vx1;
    float w10 = ly         * (1.f - lx) * vy1 * vx0;
    float w11 = ly         * lx         * vy1 * vx1;
    int i00 = yc0 * WW + xc0, i01 = yc0 * WW + xc1;
    int i10 = yc1 * WW + xc0, i11 = yc1 * WW + xc1;
    const float* xc0p = xb + (size_t)(cg * 32) * HW;
    #pragma unroll
    for (int i = 0; i < 32; i += 2) {
        const float* p0 = xc0p + (size_t)i * HW;
        const float* p1 = p0 + HW;
        float s0 = w00 * p0[i00] + w01 * p0[i01] + w10 * p0[i10] + w11 * p0[i11];
        float s1 = w00 * p1[i00] + w01 * p1[i01] + w10 * p1[i10] + w11 * p1[i11];
        *(unsigned*)&buf[p * BSTR + cg * 32 + i] = pack_bf16(s0, s1);
    }
}

// -------- fused: offset conv (WMMA) + bilinear im2col + deform GEMM (WMMA) ---
template <bool NHWC>
__global__ __launch_bounds__(256)
void deform_fused(const float* __restrict__ x,
                  const unsigned short* __restrict__ xh,
                  const unsigned short* __restrict__ w_main,
                  const unsigned short* __restrict__ w_offp,
                  float* __restrict__ out) {
    const int ho   = blockIdx.x;
    const int b    = blockIdx.y;
    const int tid  = threadIdx.x;
    const int lane = tid & 31;
    const int wv   = tid >> 5;
    const int m    = lane & 15;
    const int half = lane >> 4;

    __shared__ __align__(16) unsigned short s_bx[2][64 * BSTR];
    __shared__ float s_off[32 * OSTR];

    const float*          xb  = x  + (size_t)b * CIN * HW;
    const unsigned short* xhb = xh + (size_t)b * HW * CIN;

    const int cg = tid >> 6;
    const int p  = tid & 63;

    // =========== Phase 1: offset conv, M=32(pad 18) x K=1152 x N=64 ==========
    {
        const int mstrip = wv >> 2;
        const int ntile  = wv & 3;
        v8f acc = {0.f,0.f,0.f,0.f,0.f,0.f,0.f,0.f};
        if constexpr (NHWC) stage_int_h(s_bx[0], xhb, ho, p, cg, 0, 0);
        else                stage_int_f(s_bx[0], xb,  ho, p, cg, 0, 0);
        __syncthreads();
        int kyn = 0, kxn = 1;
        for (int kk = 0; kk < KK; ++kk) {
            const unsigned short* bb = s_bx[kk & 1];
            const unsigned short* wk = w_offp + kk * 32 * CIN;
            #pragma unroll
            for (int kc = 0; kc < 4; ++kc) {
                const int cbA = kc * 32 + (half << 3);
                Frag A;
                A.q[0] = *(const uint4*)&wk[(mstrip * 16 + m) * CIN + cbA];
                A.q[1] = *(const uint4*)&wk[(mstrip * 16 + m) * CIN + cbA + 16];
                const int cbB = kc * 32 + (half << 3);
                Frag Bf;
                Bf.q[0] = *(const uint4*)&bb[(ntile * 16 + m) * BSTR + cbB];
                Bf.q[1] = *(const uint4*)&bb[(ntile * 16 + m) * BSTR + cbB + 16];
                acc = __builtin_amdgcn_wmma_f32_16x16x32_bf16(
                        false, A.v, false, Bf.v, (short)0, acc, false, false);
            }
            if (kk < KK - 1) {
                if constexpr (NHWC) stage_int_h(s_bx[(kk + 1) & 1], xhb, ho, p, cg, kyn, kxn);
                else                stage_int_f(s_bx[(kk + 1) & 1], xb,  ho, p, cg, kyn, kxn);
                kxn = (kxn == 2) ? 0 : kxn + 1;
                if (kxn == 0) kyn++;
            }
            __syncthreads();
        }
        const int nloc = ntile * 16 + m;
        const int mrow = mstrip * 16 + (half << 3);
        #pragma unroll
        for (int v = 0; v < 8; ++v)
            s_off[(mrow + v) * OSTR + nloc] = acc[v];   // rows 18..31 = padding
    }
    __syncthreads();

    // =========== Phase 2: bilinear im2col -> WMMA GEMM 128x1152x64 ===========
    v8f acc[4];
    #pragma unroll
    for (int t = 0; t < 4; ++t) acc[t] = (v8f){0.f,0.f,0.f,0.f,0.f,0.f,0.f,0.f};

    {
        if constexpr (NHWC) stage_bilin_h(s_bx[0], xhb, s_off, ho, p, cg, 0, 0, 0);
        else                stage_bilin_f(s_bx[0], xb,  s_off, ho, p, cg, 0, 0, 0);
        __syncthreads();
        int kyn = 0, kxn = 1;
        for (int kk = 0; kk < KK; ++kk) {
            const unsigned short* bb = s_bx[kk & 1];
            const unsigned short* wk = w_main + kk * COUT * CIN;
            #pragma unroll
            for (int kc = 0; kc < 4; ++kc) {
                const int cbA = kc * 32 + (half << 3);
                Frag A;
                A.q[0] = *(const uint4*)&wk[(wv * 16 + m) * CIN + cbA];
                A.q[1] = *(const uint4*)&wk[(wv * 16 + m) * CIN + cbA + 16];
                const int cbB = kc * 32 + (half << 3);
                #pragma unroll
                for (int t = 0; t < 4; ++t) {
                    Frag Bf;
                    Bf.q[0] = *(const uint4*)&bb[(t * 16 + m) * BSTR + cbB];
                    Bf.q[1] = *(const uint4*)&bb[(t * 16 + m) * BSTR + cbB + 16];
                    acc[t] = __builtin_amdgcn_wmma_f32_16x16x32_bf16(
                               false, A.v, false, Bf.v, (short)0, acc[t], false, false);
                }
            }
            if (kk < KK - 1) {
                if constexpr (NHWC)
                    stage_bilin_h(s_bx[(kk + 1) & 1], xhb, s_off, ho, p, cg, kk + 1, kyn, kxn);
                else
                    stage_bilin_f(s_bx[(kk + 1) & 1], xb,  s_off, ho, p, cg, kk + 1, kyn, kxn);
                kxn = (kxn == 2) ? 0 : kxn + 1;
                if (kxn == 0) kyn++;
            }
            __syncthreads();
        }
    }

    // =========== epilogue ===========
    #pragma unroll
    for (int t = 0; t < 4; ++t) {
        #pragma unroll
        for (int v = 0; v < 8; ++v) {
            int co = wv * 16 + (half << 3) + v;
            int wo = t * 16 + m;
            out[(((size_t)b * COUT + co) * HH + ho) * WW + wo] = acc[t][v];
        }
    }
}

extern "C" void kernel_launch(void* const* d_in, const int* in_sizes, int n_in,
                              void* d_out, int out_size, void* d_ws, size_t ws_size,
                              hipStream_t stream) {
    const float* x    = (const float*)d_in[0];  // [8][128][64][64]
    const float* woff = (const float*)d_in[1];  // [18][128][3][3]
    const float* w    = (const float*)d_in[2];  // [128][128][3][3]
    float* out = (float*)d_out;

    const size_t nxh = (size_t)BATCH * HH * WW * CIN;
    const size_t nwm = (size_t)KK * COUT * CIN;
    const size_t nwo = (size_t)KK * 32 * CIN;
    dim3 grid(HH, BATCH);

    if (ws_size >= (nxh + nwm + nwo) * sizeof(unsigned short)) {
        unsigned short* xh     = (unsigned short*)d_ws;
        unsigned short* w_main = xh + nxh;
        unsigned short* w_offp = w_main + nwm;
        prep_w<<<(int)((nwm + 255) / 256), 256, 0, stream>>>(w, woff, w_main, w_offp);
        prep_x<<<dim3(HH, BATCH), 256, 0, stream>>>(x, xh);
        deform_fused<true><<<grid, 256, 0, stream>>>(x, xh, w_main, w_offp, out);
    } else {
        unsigned short* w_main = (unsigned short*)d_ws;
        unsigned short* w_offp = w_main + nwm;
        prep_w<<<(int)((nwm + 255) / 256), 256, 0, stream>>>(w, woff, w_main, w_offp);
        deform_fused<false><<<grid, 256, 0, stream>>>(x, (const unsigned short*)nullptr,
                                                      w_main, w_offp, out);
    }
}
